// TtMambaSSM_37074157699751
// MI455X (gfx1250) — matrix operand order repaired and checked by reading, not compiled, not run on hardware
//
#include <hip/hip_runtime.h>

typedef __bf16 v16bf __attribute__((ext_vector_type(16)));
typedef float  v8f   __attribute__((ext_vector_type(8)));
typedef float  v4f   __attribute__((ext_vector_type(4)));

#define D_INNER 5120
#define DT_RANK 160
#define N_STATE 16
#define BATCH   256

// ---------------------------------------------------------------------------
// Load one 16x32 bf16 A-fragment / 32x16 bf16 B-fragment for a K-step at k0.
// Contiguous k order: element s of lane half h holds k = k0 + 16*h + s (A and B alike).
// ---------------------------------------------------------------------------
__device__ __forceinline__ v16bf load_a_frag(const float* __restrict__ arow, int k0)
{
    v16bf a;
    #pragma unroll
    for (int e = 0; e < 8; ++e) {
        a[e]     = (__bf16)arow[k0 + e];        // K = k0 + 16*half + e
        a[e + 8] = (__bf16)arow[k0 + 8 + e];    // K = k0 + 16*half + 8 + e
    }
    return a;
}

__device__ __forceinline__ v16bf load_b_frag(const float* __restrict__ bcol, int k0, int N)
{
    v16bf b;
    #pragma unroll
    for (int v = 0; v < 8; ++v) {
        b[2 * v]     = (__bf16)bcol[(size_t)(k0 + 2 * v) * N];
        b[2 * v + 1] = (__bf16)bcol[(size_t)(k0 + 2 * v + 1) * N];
    }
    return b;
}

// ---------------------------------------------------------------------------
// Row-major WMMA GEMM with on-the-fly f32->bf16 operand conversion:
//   C[M,N] = A[M,K] @ B[K,N]   (lda=K, ldb=ldc=N), f32 accumulate.
// One 16x16 output tile per wave via v_wmma_f32_16x16x32_bf16 (K-step 32);
// 2x unrolled with independent accumulators to break the wmma->wmma RAW chain.
// 8 waves per 256-thread block. K must be a multiple of 32.
// mode 0: plain store.  mode 1: += bias[col], then softplus (threshold 20).
// ---------------------------------------------------------------------------
__global__ __launch_bounds__(256)
void gemm_wmma_bf16(const float* __restrict__ A, const float* __restrict__ B,
                    float* __restrict__ C, const float* __restrict__ bias,
                    int M, int N, int K, int mode)
{
    const int lane = threadIdx.x & 31;
    const int wave = threadIdx.x >> 5;

    const int nTilesN = N >> 4;
    const int nTiles  = (M >> 4) * nTilesN;
    const int tile    = blockIdx.x * 8 + wave;
    if (tile >= nTiles) return;            // whole-wave uniform exit

    const int mt = tile / nTilesN;
    const int nt = tile - mt * nTilesN;
    const int m0 = mt << 4;
    const int n0 = nt << 4;

    const int half = lane >> 4;            // 0: lanes 0-15, 1: lanes 16-31
    const int l    = lane & 15;

    // A row for this lane, pre-offset by the half-dependent K base (16*half).
    const float* __restrict__ arow = A + (size_t)(m0 + l) * K + 16 * half;
    // B column for this lane, pre-offset by half-dependent K base (16*half rows).
    const float* __restrict__ bcol = B + (size_t)(16 * half) * N + (n0 + l);

    v8f c0 = {};                           // two independent f32 accumulators
    v8f c1 = {};

    int k0 = 0;
    for (; k0 + 64 <= K; k0 += 64) {
        v16bf a0 = load_a_frag(arow, k0);
        v16bf b0 = load_b_frag(bcol, k0, N);
        v16bf a1 = load_a_frag(arow, k0 + 32);
        v16bf b1 = load_b_frag(bcol, k0 + 32, N);
        c0 = __builtin_amdgcn_wmma_f32_16x16x32_bf16(
                false, a0, false, b0, (short)0, c0, false, false);
        c1 = __builtin_amdgcn_wmma_f32_16x16x32_bf16(
                false, a1, false, b1, (short)0, c1, false, false);
    }
    for (; k0 < K; k0 += 32) {             // K % 64 remainder (e.g. K=160)
        v16bf a0 = load_a_frag(arow, k0);
        v16bf b0 = load_b_frag(bcol, k0, N);
        c0 = __builtin_amdgcn_wmma_f32_16x16x32_bf16(
                false, a0, false, b0, (short)0, c0, false, false);
    }

    // C/D layout: VGPR i -> row m0 + i + 8*half, col n0 + lane%16
    const int col = n0 + l;
    #pragma unroll
    for (int i = 0; i < 8; ++i) {
        float v = c0[i] + c1[i];
        if (mode == 1) {
            v += bias[col];
            // softplus with linear region above 20 (matches reference)
            v = (v > 20.0f) ? v : __logf(1.0f + __expf(fminf(v, 20.0f)));
        }
        C[(size_t)(m0 + i + 8 * half) * N + col] = v;
    }
}

// ---------------------------------------------------------------------------
// State update: per (b,d) thread.
//   abar = exp(delta*A[d,:]); h = abar*h0 + delta*Bp[b,:]*x
//   y = sum_n h*Cp[b,n];  out = x*D[d] + y
// Dominant traffic: h0 (84 MB, read-once) -> non-temporal global_load_b128 so
// the streamed state does not evict reused operands from WGP$/L2.
// One batch row per block -> Bp/Cp loads broadcast within the block.
// ---------------------------------------------------------------------------
__global__ __launch_bounds__(256)
void ssm_update(const float* __restrict__ x, const float* __restrict__ delta,
                const float* __restrict__ Bp, const float* __restrict__ Cp,
                const float* __restrict__ Amat, const float* __restrict__ Dvec,
                const float* __restrict__ h0, float* __restrict__ out)
{
    const int idx = blockIdx.x * blockDim.x + threadIdx.x;  // b*D_INNER + d
    const int b   = idx / D_INNER;
    const int d   = idx - b * D_INNER;

    const float xv = x[idx];
    const float dl = delta[idx];

    const v4f* __restrict__ Ap = (const v4f*)(Amat + (size_t)d   * N_STATE);
    const v4f* __restrict__ Hp = (const v4f*)(h0   + (size_t)idx * N_STATE);
    const v4f* __restrict__ Bq = (const v4f*)(Bp   + (size_t)b   * N_STATE);
    const v4f* __restrict__ Cq = (const v4f*)(Cp   + (size_t)b   * N_STATE);

    float y = 0.0f;
    #pragma unroll
    for (int q = 0; q < 4; ++q) {
        const v4f a  = Ap[q];
        const v4f h  = __builtin_nontemporal_load(&Hp[q]);  // read-once stream
        const v4f Bv = Bq[q];
        const v4f Cv = Cq[q];
        y += (__expf(dl * a.x) * h.x + dl * Bv.x * xv) * Cv.x;
        y += (__expf(dl * a.y) * h.y + dl * Bv.y * xv) * Cv.y;
        y += (__expf(dl * a.z) * h.z + dl * Bv.z * xv) * Cv.z;
        y += (__expf(dl * a.w) * h.w + dl * Bv.w * xv) * Cv.w;
    }
    __builtin_nontemporal_store(xv * Dvec[d] + y, &out[idx]);  // write-once stream
}

extern "C" void kernel_launch(void* const* d_in, const int* in_sizes, int n_in,
                              void* d_out, int out_size, void* d_ws, size_t ws_size,
                              hipStream_t stream)
{
    (void)in_sizes; (void)n_in; (void)out_size; (void)ws_size;

    const float* x         = (const float*)d_in[0];  // (256, 5120)
    const float* W_dt_low  = (const float*)d_in[1];  // (5120, 160)
    const float* W_dt      = (const float*)d_in[2];  // (160, 5120)
    const float* b_dt      = (const float*)d_in[3];  // (5120,)
    const float* W_B       = (const float*)d_in[4];  // (5120, 16)
    const float* W_C       = (const float*)d_in[5];  // (5120, 16)
    const float* Amat      = (const float*)d_in[6];  // (5120, 16)
    const float* Dvec      = (const float*)d_in[7];  // (5120,)
    const float* h0        = (const float*)d_in[8];  // (256, 5120, 16)
    float*       out       = (float*)d_out;          // (256, 5120)

    // Workspace carve-out (floats): T | delta | Bp | Cp  (~5.45 MB total)
    float* ws    = (float*)d_ws;
    float* T     = ws;                                   // 256*160
    float* delta = T + (size_t)BATCH * DT_RANK;          // 256*5120
    float* Bpv   = delta + (size_t)BATCH * D_INNER;      // 256*16
    float* Cpv   = Bpv + (size_t)BATCH * N_STATE;        // 256*16

    // 1) T = x @ W_dt_low              M=256, N=160, K=5120
    {
        const int tiles = (BATCH / 16) * (DT_RANK / 16);     // 160
        gemm_wmma_bf16<<<(tiles + 7) / 8, 256, 0, stream>>>(
            x, W_dt_low, T, nullptr, BATCH, DT_RANK, D_INNER, 0);
    }
    // 2) delta = softplus(T @ W_dt + b_dt)   M=256, N=5120, K=160
    {
        const int tiles = (BATCH / 16) * (D_INNER / 16);     // 5120
        gemm_wmma_bf16<<<(tiles + 7) / 8, 256, 0, stream>>>(
            T, W_dt, delta, b_dt, BATCH, D_INNER, DT_RANK, 1);
    }
    // 3) Bp = x @ W_B ; Cp = x @ W_C   M=256, N=16, K=5120
    {
        const int tiles = (BATCH / 16) * (N_STATE / 16);     // 16
        gemm_wmma_bf16<<<(tiles + 7) / 8, 256, 0, stream>>>(
            x, W_B, Bpv, nullptr, BATCH, N_STATE, D_INNER, 0);
        gemm_wmma_bf16<<<(tiles + 7) / 8, 256, 0, stream>>>(
            x, W_C, Cpv, nullptr, BATCH, N_STATE, D_INNER, 0);
    }
    // 4) State update + output (bandwidth-dominated: streams h0 once)
    ssm_update<<<(BATCH * D_INNER) / 256, 256, 0, stream>>>(
        x, delta, Bpv, Cpv, Amat, Dvec, h0, out);
}
